// SelfAttentionBlock_59889023975808
// MI455X (gfx1250) — matrix operand order repaired and checked by reading, not compiled, not run on hardware
//
#include <hip/hip_runtime.h>
#include <hip/hip_bf16.h>

// ---------------------------------------------------------------------------
// SelfAttentionBlock for MI455X (gfx1250).
//  * all GEMMs + attention on v_wmma_f32_16x16x32_f16 (f32 accumulate)
//  * f16 operands staged with GLOBAL_LOAD_ASYNC_TO_LDS_B128 (ASYNCcnt),
//    double-buffered so copies of tile i+1 overlap WMMAs of tile i
//  * template-specialized branch-free epilogues
// ---------------------------------------------------------------------------

#define BQ   4
#define NQ   2048
#define DQ   512
#define HQ   8
#define HDQ  64
#define MROWS (BQ * NQ)        // 8192

typedef __attribute__((ext_vector_type(16))) _Float16 v16h;
typedef __attribute__((ext_vector_type(8)))  float    v8f;
typedef __attribute__((ext_vector_type(4)))  int      v4i;

union FragA { unsigned int u[8]; v16h v; };
union FragB { uint4 q[2];        v16h v; };
union H8    { uint4 q;           _Float16 h[8]; };

#if defined(__HIP_DEVICE_COMPILE__) && __has_builtin(__builtin_amdgcn_global_load_async_to_lds_b128)
#define USE_ASYNC_LDS 1
#else
#define USE_ASYNC_LDS 0
#endif

#if USE_ASYNC_LDS
typedef __attribute__((address_space(1))) v4i* gas_v4i;
typedef __attribute__((address_space(3))) v4i* las_v4i;
#endif

// copy 16 bytes global -> LDS (async path tracked by ASYNCcnt)
__device__ __forceinline__ void cp16_g2l(const _Float16* g, _Float16* l) {
#if USE_ASYNC_LDS
  __builtin_amdgcn_global_load_async_to_lds_b128(
      (gas_v4i)(unsigned long long)g,   // global VA == flat VA on AMDGPU
      (las_v4i)l,                       // generic->AS3 addrspacecast
      0, 0);
#else
  *reinterpret_cast<uint4*>(l) = *reinterpret_cast<const uint4*>(g);
#endif
}

__device__ __forceinline__ void wait_async_lds() {
#if USE_ASYNC_LDS
  asm volatile("s_wait_asynccnt 0x0" ::: "memory");
#endif
}

__device__ __forceinline__ v8f wmma_f16(v16h a, v16h b, v8f c) {
  return __builtin_amdgcn_wmma_f32_16x16x32_f16(
      false, a, false, b, (short)0, c, false, false);
}

// A fragment (16x32 f16), contiguous k = hi*16 + s order. tile -> row 0 of the window.
__device__ __forceinline__ v16h ld_frag_a(const _Float16* tile, int ldh, int lane) {
  int m  = lane & 15;
  int hi = (lane >> 4) & 1;
  const unsigned int* row = reinterpret_cast<const unsigned int*>(tile + m * ldh);
  FragA f;
#pragma unroll
  for (int v = 0; v < 4; ++v) f.u[v]     = row[hi * 8 + v];      // K = hi*16 + 2v
#pragma unroll
  for (int v = 0; v < 4; ++v) f.u[4 + v] = row[hi * 8 + 4 + v];  // K = hi*16 + 8 + 2v
  return f.v;
}

// B fragment (32x16 f16) from transposed tile Bt[n][k]: 2x ds_load_b128.
__device__ __forceinline__ v16h ld_frag_b(const _Float16* bt, int ldh, int k0, int lane) {
  int n  = lane & 15;
  int hi = (lane >> 4) & 1;
  const uint4* p = reinterpret_cast<const uint4*>(bt + n * ldh + k0 + hi * 16);
  FragB f;
  f.q[0] = p[0];
  f.q[1] = p[1];
  return f.v;
}

// ---------------------------------------------------------------------------
// elementwise converters
// ---------------------------------------------------------------------------
__global__ void cvt_f16_kernel(const float* __restrict__ in,
                               _Float16* __restrict__ out, int n) {
  int i = blockIdx.x * blockDim.x + threadIdx.x;
  if (i < n) out[i] = (_Float16)in[i];
}

// W (K x N, row-major f32) -> Wt (N x K, row-major f16)
__global__ void transpose_cvt_kernel(const float* __restrict__ in,
                                     _Float16* __restrict__ out, int K, int N) {
  int i = blockIdx.x * blockDim.x + threadIdx.x;
  if (i >= K * N) return;
  int k = i / N, n = i - k * N;
  out[(size_t)n * K + k] = (_Float16)in[i];
}

// ---------------------------------------------------------------------------
// GEMM: C[M,N] = concat(A0|A1)[M,K](f16) @ Bt^T + bias (+residual)
// Bt is N-major f16 (N x K). Block 128x128, BK=32, 8 waves, wave tile 32x64.
// Double-buffered async staging: copies of tile i+1 overlap WMMAs of tile i.
// ---------------------------------------------------------------------------
#define GBM 128
#define GBN 128
#define GBK 32
#define GLD 40   // padded LDS row stride in halves (80B = 5*16B, keeps align)

template<bool CONCAT, bool RESID, bool F16OUT>
__global__ __launch_bounds__(256, 2)
void gemm_f16_kernel(const _Float16* __restrict__ A0, const _Float16* __restrict__ A1,
                     int ksplit,
                     const _Float16* __restrict__ Bt, const float* __restrict__ bias,
                     const float* __restrict__ residual,
                     float* __restrict__ Cf, _Float16* __restrict__ Ch,
                     int M, int N, int K) {
  __shared__ __align__(16) _Float16 As[2][GBM * GLD];
  __shared__ __align__(16) _Float16 Bts[2][GBN * GLD];

  const int tid  = threadIdx.x;
  const int lane = tid & 31;
  const int wave = tid >> 5;
  const int wr   = wave & 3;
  const int wc   = wave >> 2;
  const int m0   = blockIdx.y * GBM;
  const int n0   = blockIdx.x * GBN;

  // stage one 128x32 A tile + 128x32 B tile into buffer `buf` (4 cp16/thread)
  auto stage = [&](int buf, int k0) {
#pragma unroll
    for (int it = 0; it < 2; ++it) {
      int idx = tid + it * 256;
      int r   = idx >> 2;
      int ch  = (idx & 3) * 8;       // halves within row
      int gk  = k0 + ch;
      const _Float16* src;
      if (CONCAT && gk >= ksplit)
        src = A1 + (size_t)(m0 + r) * (size_t)(K - ksplit) + (gk - ksplit);
      else
        src = A0 + (size_t)(m0 + r) * (size_t)ksplit + gk;
      cp16_g2l(src, &As[buf][r * GLD + ch]);
    }
#pragma unroll
    for (int it = 0; it < 2; ++it) {
      int idx = tid + it * 256;
      int r   = idx >> 2;
      int ch  = (idx & 3) * 8;
      cp16_g2l(Bt + (size_t)(n0 + r) * (size_t)K + k0 + ch, &Bts[buf][r * GLD + ch]);
    }
  };

  v8f acc[2][4];
#pragma unroll
  for (int mt = 0; mt < 2; ++mt)
#pragma unroll
    for (int nt = 0; nt < 4; ++nt)
      acc[mt][nt] = v8f{0.f, 0.f, 0.f, 0.f, 0.f, 0.f, 0.f, 0.f};

  const int nk = K / GBK;
  stage(0, 0);
  for (int ki = 0; ki < nk; ++ki) {
    const int cur = ki & 1;
    wait_async_lds();        // this wave's copies for tile ki complete
    __syncthreads();         // all waves' copies done; prior reads retired
    if (ki + 1 < nk) stage(cur ^ 1, (ki + 1) * GBK);   // overlap with compute

    v16h af0 = ld_frag_a(&As[cur][(wr * 32 +  0) * GLD], GLD, lane);
    v16h af1 = ld_frag_a(&As[cur][(wr * 32 + 16) * GLD], GLD, lane);
#pragma unroll
    for (int nt = 0; nt < 4; ++nt) {
      v16h bf = ld_frag_b(&Bts[cur][(wc * 64 + nt * 16) * GLD], GLD, 0, lane);
      acc[0][nt] = wmma_f16(af0, bf, acc[0][nt]);
      acc[1][nt] = wmma_f16(af1, bf, acc[1][nt]);
    }
  }

  // ---- epilogue (branch-free): row = j + (lane>=16)*8, col = lane&15
  const int hi = lane >> 4;
  const int nl = lane & 15;
#pragma unroll
  for (int mt = 0; mt < 2; ++mt)
#pragma unroll
    for (int nt = 0; nt < 4; ++nt) {
      int col = n0 + wc * 64 + nt * 16 + nl;
      float bv = bias[col];
#pragma unroll
      for (int j = 0; j < 8; ++j) {
        size_t off = (size_t)(m0 + wr * 32 + mt * 16 + j + hi * 8) * (size_t)N + col;
        float v = acc[mt][nt][j] + bv;
        if (RESID) v += residual[off];
        if (F16OUT) Ch[off] = (_Float16)v;
        else        Cf[off] = v;
      }
    }
}

// ---------------------------------------------------------------------------
// RoPE in place on f16 qkv (B,N,1536)
// ---------------------------------------------------------------------------
__global__ void rope_kernel(_Float16* __restrict__ qkv, const float* __restrict__ freqs) {
  int idx = blockIdx.x * blockDim.x + threadIdx.x;   // B*N*H*32 pairs
  if (idx >= BQ * NQ * HQ * (HDQ / 2)) return;
  int i  = idx & 31;
  int h  = (idx >> 5) & 7;
  int bn = idx >> 8;
  float f = freqs[(size_t)bn * 32 + i];
  float c = cosf(f), s = sinf(f);
  _Float16* row = qkv + (size_t)bn * (3 * DQ);
  int cq = h * HDQ + 2 * i;
  float x1 = (float)row[cq], x2 = (float)row[cq + 1];
  row[cq]     = (_Float16)(x1 * c - x2 * s);
  row[cq + 1] = (_Float16)(x1 * s + x2 * c);
  int ck = DQ + cq;
  float k1 = (float)row[ck], k2 = (float)row[ck + 1];
  row[ck]     = (_Float16)(k1 * c - k2 * s);
  row[ck + 1] = (_Float16)(k1 * s + k2 * c);
}

// ---------------------------------------------------------------------------
// Flash attention: grid (N/64, B*H), 128 threads (4 waves x 16 q-rows).
// Double-buffered K/V tiles; qkv f16 -> msg f16 (B,N,D)
// ---------------------------------------------------------------------------
__global__ __launch_bounds__(128, 2)
void flash_attn_kernel(const _Float16* __restrict__ qkv, _Float16* __restrict__ msg) {
  __shared__ __align__(16) _Float16 Kt[2][64 * HDQ];  // [key][hd]
  __shared__ __align__(16) _Float16 Vt[2][HDQ * 64];  // [hd][key]
  __shared__ __align__(16) _Float16 Pw[4][16 * HDQ];  // per-wave Q/P scratch

  const int tid  = threadIdx.x;
  const int lane = tid & 31;
  const int wave = tid >> 5;
  const int b    = blockIdx.y >> 3;
  const int h    = blockIdx.y & 7;
  const int qblk = blockIdx.x * 64;
  const int hi   = lane >> 4;
  const int nl   = lane & 15;
  const float scale = 0.125f;   // 64^-0.5

  // stage K tile (async, row-major) + V tile (manual transpose) for key tile kt
  auto stageKV = [&](int buf, int kt) {
#pragma unroll
    for (int it = 0; it < 4; ++it) {
      int idx = tid + it * 128;
      int r   = idx >> 3;
      int ch  = (idx & 7) * 8;
      cp16_g2l(qkv + ((size_t)(b * NQ + kt * 64 + r)) * (3 * DQ) + DQ + h * HDQ + ch,
               &Kt[buf][r * HDQ + ch]);
    }
#pragma unroll
    for (int it = 0; it < 4; ++it) {
      int idx = tid + it * 128;
      int r   = idx >> 3;
      int c8  = (idx & 7) * 8;
      H8 v;
      v.q = *reinterpret_cast<const uint4*>(
          qkv + ((size_t)(b * NQ + kt * 64 + r)) * (3 * DQ) + 2 * DQ + h * HDQ + c8);
#pragma unroll
      for (int i = 0; i < 8; ++i) Vt[buf][(c8 + i) * 64 + r] = v.h[i];
    }
  };

  // ---- stage this wave's 16 Q rows (async), build A fragments
  _Float16* myP = &Pw[wave][0];
#pragma unroll
  for (int it = 0; it < 2; ++it) {
    int idx = lane + it * 32;            // 16 rows x 4 chunks
    int r   = idx >> 2;
    int ch  = (idx & 3) * 8;
    const _Float16* src = qkv + ((size_t)(b * NQ + qblk + wave * 16 + r)) * (3 * DQ)
                              + h * HDQ + ch;
    cp16_g2l(src, myP + r * HDQ + ch);
  }
  stageKV(0, 0);               // first K/V tile overlaps Q fragment build
  wait_async_lds();
  __syncthreads();
  v16h qf0 = ld_frag_a(myP,      HDQ, lane);
  v16h qf1 = ld_frag_a(myP + 32, HDQ, lane);

  float mrow[8], lrow[8];
  v8f o[4];
#pragma unroll
  for (int j = 0; j < 8; ++j) { mrow[j] = -1e30f; lrow[j] = 0.f; }
#pragma unroll
  for (int t = 0; t < 4; ++t) o[t] = v8f{0.f,0.f,0.f,0.f,0.f,0.f,0.f,0.f};

  const int nkt = NQ / 64;
  for (int kt = 0; kt < nkt; ++kt) {
    const int cur = kt & 1;
    wait_async_lds();
    __syncthreads();
    if (kt + 1 < nkt) stageKV(cur ^ 1, kt + 1);    // overlap with compute

    // ---- S = Q K^T
    v8f s[4];
#pragma unroll
    for (int nt = 0; nt < 4; ++nt) {
      v8f z = v8f{0.f,0.f,0.f,0.f,0.f,0.f,0.f,0.f};
      z = wmma_f16(qf0, ld_frag_b(&Kt[cur][(nt * 16) * HDQ], HDQ,  0, lane), z);
      z = wmma_f16(qf1, ld_frag_b(&Kt[cur][(nt * 16) * HDQ], HDQ, 32, lane), z);
      s[nt] = z;
    }

    // ---- online softmax (rows striped j + hi*8; reduce within 16-lane half)
#pragma unroll
    for (int j = 0; j < 8; ++j) {
      float rm = -1e30f;
#pragma unroll
      for (int nt = 0; nt < 4; ++nt) rm = fmaxf(rm, s[nt][j] * scale);
#pragma unroll
      for (int off = 1; off < 16; off <<= 1) rm = fmaxf(rm, __shfl_xor(rm, off, 32));
      float mn    = fmaxf(mrow[j], rm);
      float alpha = __expf(mrow[j] - mn);
      mrow[j] = mn;
      float rs = 0.f;
#pragma unroll
      for (int nt = 0; nt < 4; ++nt) {
        float p = __expf(s[nt][j] * scale - mn);
        myP[(j + hi * 8) * HDQ + nt * 16 + nl] = (_Float16)p;
        rs += p;
      }
#pragma unroll
      for (int off = 1; off < 16; off <<= 1) rs += __shfl_xor(rs, off, 32);
      lrow[j] = lrow[j] * alpha + rs;
#pragma unroll
      for (int t = 0; t < 4; ++t) o[t][j] *= alpha;
    }

    // ---- O += P V
    v16h pf0 = ld_frag_a(myP,      HDQ, lane);
    v16h pf1 = ld_frag_a(myP + 32, HDQ, lane);
#pragma unroll
    for (int t = 0; t < 4; ++t) {
      o[t] = wmma_f16(pf0, ld_frag_b(&Vt[cur][(t * 16) * 64], 64,  0, lane), o[t]);
      o[t] = wmma_f16(pf1, ld_frag_b(&Vt[cur][(t * 16) * 64], 64, 32, lane), o[t]);
    }
  }

  // ---- normalize + write msg (f16)
#pragma unroll
  for (int t = 0; t < 4; ++t)
#pragma unroll
    for (int j = 0; j < 8; ++j) {
      int row = qblk + wave * 16 + j + hi * 8;
      int col = h * HDQ + t * 16 + nl;
      msg[((size_t)(b * NQ) + row) * DQ + col] = (_Float16)(o[t][j] / lrow[j]);
    }
}

// ---------------------------------------------------------------------------
// LayerNorm (biased var) + exact GELU; reads f32 hcat, writes f16 hcath
// ---------------------------------------------------------------------------
__global__ __launch_bounds__(256)
void ln_gelu_kernel(const float* __restrict__ hcat, _Float16* __restrict__ hcath,
                    const float* __restrict__ g, const float* __restrict__ be) {
  __shared__ float ss[256];
  __shared__ float sq[256];
  const int tid = threadIdx.x;
  const float* p = hcat + (size_t)blockIdx.x * 1024;
  _Float16* ph = hcath + (size_t)blockIdx.x * 1024;
  float s = 0.f, q = 0.f;
#pragma unroll
  for (int i = 0; i < 4; ++i) {
    float v = p[tid + i * 256];
    s += v; q += v * v;
  }
  ss[tid] = s; sq[tid] = q;
  __syncthreads();
  for (int off = 128; off > 0; off >>= 1) {
    if (tid < off) { ss[tid] += ss[tid + off]; sq[tid] += sq[tid + off]; }
    __syncthreads();
  }
  float mean = ss[0] * (1.f / 1024.f);
  float var  = sq[0] * (1.f / 1024.f) - mean * mean;
  float rstd = rsqrtf(var + 1e-5f);
#pragma unroll
  for (int i = 0; i < 4; ++i) {
    int c = tid + i * 256;
    float y = (p[c] - mean) * rstd * g[c] + be[c];
    ph[c] = (_Float16)(0.5f * y * (1.f + erff(y * 0.70710678118654752f)));
  }
}

// ---------------------------------------------------------------------------
extern "C" void kernel_launch(void* const* d_in, const int* in_sizes, int n_in,
                              void* d_out, int out_size, void* d_ws, size_t ws_size,
                              hipStream_t stream) {
  (void)in_sizes; (void)n_in; (void)out_size; (void)ws_size;
  const float* x      = (const float*)d_in[0];
  const float* freqs  = (const float*)d_in[1];
  const float* wqkv_w = (const float*)d_in[2];
  const float* wqkv_b = (const float*)d_in[3];
  const float* out_w  = (const float*)d_in[4];
  const float* out_b  = (const float*)d_in[5];
  const float* ffn1_w = (const float*)d_in[6];
  const float* ffn1_b = (const float*)d_in[7];
  const float* ln_g   = (const float*)d_in[8];
  const float* ln_b   = (const float*)d_in[9];
  const float* ffn2_w = (const float*)d_in[10];
  const float* ffn2_b = (const float*)d_in[11];
  float* out = (float*)d_out;

  // workspace: region0 = qkvh(8192x1536 h) + msgAh(8192x512 h); after attention
  // that exact region is reused as hcat (8192x1024 f32). Then f16 buffers.
  char* base = (char*)d_ws;
  _Float16* qkvh  = (_Float16*)base;
  _Float16* msgAh = qkvh + (size_t)MROWS * 1536;
  float*    hcat  = (float*)base;                       // overlays qkvh+msgAh
  char* p = base + (size_t)MROWS * 2048 * 2;
  _Float16* xh    = (_Float16*)p;  p += (size_t)MROWS * 512 * 2;
  _Float16* msgPh = (_Float16*)p;  p += (size_t)MROWS * 512 * 2;
  _Float16* hcath = (_Float16*)p;  p += (size_t)MROWS * 1024 * 2;
  _Float16* wqkvT = (_Float16*)p;  p += (size_t)512 * 1536 * 2;
  _Float16* outT  = (_Float16*)p;  p += (size_t)512 * 512 * 2;
  _Float16* ffn1T = (_Float16*)p;  p += (size_t)1024 * 1024 * 2;
  _Float16* ffn2T = (_Float16*)p;  p += (size_t)1024 * 512 * 2;

  // 0) one-time f16 conversions (+ weight transposes to N-major)
  cvt_f16_kernel<<<(MROWS * DQ + 255) / 256, 256, 0, stream>>>(x, xh, MROWS * DQ);
  transpose_cvt_kernel<<<(512 * 1536 + 255) / 256, 256, 0, stream>>>(wqkv_w, wqkvT, 512, 1536);
  transpose_cvt_kernel<<<(512 * 512 + 255) / 256, 256, 0, stream>>>(out_w, outT, 512, 512);
  transpose_cvt_kernel<<<(1024 * 1024 + 255) / 256, 256, 0, stream>>>(ffn1_w, ffn1T, 1024, 1024);
  transpose_cvt_kernel<<<(1024 * 512 + 255) / 256, 256, 0, stream>>>(ffn2_w, ffn2T, 1024, 512);

  // 1) qkvh = x @ wqkv_w + wqkv_b (f16 out)
  gemm_f16_kernel<false, false, true><<<dim3(1536 / GBN, MROWS / GBM), 256, 0, stream>>>(
      xh, nullptr, 512, wqkvT, wqkv_b, nullptr, nullptr, qkvh, MROWS, 1536, 512);

  // 2) RoPE in place
  rope_kernel<<<(BQ * NQ * HQ * 32 + 255) / 256, 256, 0, stream>>>(qkvh, freqs);

  // 3) flash attention -> msgAh
  flash_attn_kernel<<<dim3(NQ / 64, BQ * HQ), 128, 0, stream>>>(qkvh, msgAh);

  // 4) msgPh = msgAh @ out_w + out_b (f16 out)
  gemm_f16_kernel<false, false, true><<<dim3(512 / GBN, MROWS / GBM), 256, 0, stream>>>(
      msgAh, nullptr, 512, outT, out_b, nullptr, nullptr, msgPh, MROWS, 512, 512);

  // 5) hcat = [x, msgP] @ ffn1_w + ffn1_b (f32 out, overlays dead qkvh/msgAh)
  gemm_f16_kernel<true, false, false><<<dim3(1024 / GBN, MROWS / GBM), 256, 0, stream>>>(
      xh, msgPh, 512, ffn1T, ffn1_b, nullptr, hcat, nullptr, MROWS, 1024, 1024);

  // 6) LayerNorm + exact GELU -> hcath (f16)
  ln_gelu_kernel<<<MROWS, 256, 0, stream>>>(hcat, hcath, ln_g, ln_b);

  // 7) out = x + hcath @ ffn2_w + ffn2_b (f32 out, residual)
  gemm_f16_kernel<false, true, false><<<dim3(512 / GBN, MROWS / GBM), 256, 0, stream>>>(
      hcath, nullptr, 1024, ffn2T, ffn2_b, x, out, nullptr, MROWS, 512, 1024);
}
